// Motion_Align_35631048687928
// MI455X (gfx1250) — matrix operand order repaired and checked by reading, not compiled, not run on hardware
//
#include <hip/hip_runtime.h>

#define HH 128
#define WW 128
#define BB 4
#define NF 64
#define GG 8

typedef __attribute__((ext_vector_type(16))) __bf16 v16bf;
typedef __attribute__((ext_vector_type(8)))  float  v8f;

union Frag { v16bf v; unsigned int u[8]; uint4 q[2]; };

__device__ __forceinline__ unsigned short f2bf(float f) {
  unsigned int u = __float_as_uint(f);
  u += 0x7FFFu + ((u >> 16) & 1u);      // round-to-nearest-even
  return (unsigned short)(u >> 16);
}

__device__ __forceinline__ int iclamp(int v, int lo, int hi) {
  return v < lo ? lo : (v > hi ? hi : v);
}

// ---------------------------------------------------------------------------
// Pack fp32 weights (Cout x CIN x KK) -> bf16 (Mpad x K'), zero-pad rows,
// with K permuted to k' = r*CIN + ci (kernel-tap major, channel minor) so
// that B fragments are contiguous 16-element runs in LDS.
// ---------------------------------------------------------------------------
__global__ void pack_weights(const float* __restrict__ w, unsigned short* __restrict__ wp,
                             int Cout, int CIN, int KK, int Mpad) {
  int i = blockIdx.x * 256 + threadIdx.x;
  int K = CIN * KK;
  if (i >= Mpad * K) return;
  int row = i / K, kp = i - row * K;
  int r = kp / CIN, ci = kp - r * CIN;
  float v = (row < Cout) ? w[(size_t)row * K + ci * KK + r] : 0.f;
  wp[i] = f2bf(v);
}

// ---------------------------------------------------------------------------
// 3x3 conv (pad 1) as implicit GEMM with bf16 WMMA, fp32 accumulate.
// K' = (ky*3+kx)*CIN + ci.  LDS slab is [dy][x][ci] with padded row stride
// CINP, so each B fragment = 2 aligned ds_load_b128 at immediate offsets.
// WG = 256 threads = 8 waves = 2(M) x 4(N) tiles -> 32 Cout x 64 pixels.
// grid = (W/64, B*H, ceil(Mpad/16)/2)
// ---------------------------------------------------------------------------
template <int CIN>
__global__ void __launch_bounds__(256)
conv3x3_wmma(const float* __restrict__ in0, const float* __restrict__ in1,
             int C0,
             const unsigned short* __restrict__ Wp, const float* __restrict__ bias,
             int Cout, float* __restrict__ out, int do_lrelu) {
  constexpr int CINP = CIN + 8;                   // padded, 16B-aligned rows
  __shared__ unsigned short slds[3 * 66 * CINP];
  const int tid  = threadIdx.x;
  const int lane = tid & 31, wave = tid >> 5;
  const int lh = lane & 15, lq = lane >> 4;
  const int mw = wave >> 2, nw = wave & 3;
  const int xseg = blockIdx.x * 64;
  const int b = blockIdx.y >> 7, y = blockIdx.y & 127;
  const int co_base = (blockIdx.z * 2 + mw) * 16;

  constexpr int K = CIN * 9;                      // 576 / 1152, multiple of 32
  constexpr int Ksteps = K / 32;

  const unsigned short* Arow = Wp + (size_t)(co_base + lh) * K;
  __builtin_prefetch(Arow, 0, 1);                 // global_prefetch_b8

  // stage input slab to LDS as bf16, channel-minor: slds[(dy*66+xx)*CINP + ci]
  const int C1 = CIN - C0;
  for (int i = tid; i < CIN * 3 * 66; i += 256) {
    int xx = i % 66; int t = i / 66; int dy = t % 3; int ci = t / 3;
    int gx = xseg - 1 + xx, gy = y - 1 + dy;
    float v = 0.f;
    if (gx >= 0 && gx < WW && gy >= 0 && gy < HH) {
      const float* src = (ci < C0)
          ? in0 + ((size_t)b * C0 + ci) * (HH * WW)
          : in1 + ((size_t)b * C1 + (ci - C0)) * (HH * WW);
      v = src[gy * WW + gx];
    }
    slds[(dy * 66 + xx) * CINP + ci] = f2bf(v);
  }
  __syncthreads();

  const int xl = nw * 16 + lh;                    // local pixel column (0..63)
  const unsigned short* brow = &slds[xl * CINP + lq * 16];

  v8f acc = {0.f, 0.f, 0.f, 0.f, 0.f, 0.f, 0.f, 0.f};
#pragma unroll
  for (int kt = 0; kt < Ksteps; ++kt) {
    Frag af, bfr;
    const int kb0 = kt * 32;
#pragma unroll
    for (int i = 0; i < 8; ++i) {
      // A (16x32 bf16): VGPR i holds K = 16*lq + 2*i, +1
      int ka = kb0 + lq * 16 + i * 2;
      af.u[i] = *(const unsigned int*)(Arow + ka);
    }
    // B: within this step r = kb0/CIN is constant -> (dy,kx) constant,
    // ci spans 32 consecutive channels -> 16 contiguous ushorts per lane.
    const int r   = kb0 / CIN;
    const int rdy = r / 3, rkx = r % 3;
    const int cib = kb0 - r * CIN;
    const unsigned short* bp = brow + (rdy * 66 + rkx) * CINP + cib;
    bfr.q[0] = *(const uint4*)(bp);
    bfr.q[1] = *(const uint4*)(bp + 8);
    acc = __builtin_amdgcn_wmma_f32_16x16x32_bf16(false, af.v, false, bfr.v,
                                                  (short)0, acc, false, false);
  }

  const int colx = xseg + xl;
#pragma unroll
  for (int r = 0; r < 8; ++r) {
    int co = co_base + r + lq * 8;
    if (co < Cout) {
      float v = acc[r] + bias[co];
      if (do_lrelu) v = (v >= 0.f) ? v : 0.1f * v;
      out[(((size_t)b * Cout + co) * HH + y) * WW + colx] = v;
    }
  }
}

// ---------------------------------------------------------------------------
// Fused modulated deformable conv (DCNv2): bilinear sample+modulate to LDS
// (pixel-major, k' = kk*64 + ci minor, one ds_store_b128 per 8 channels),
// then K=576 WMMA GEMM, 18 fully-unrolled steps with b128 B loads.
// WG = 8 waves = 4(M) x 2(N) -> 64 Cout x 32 pixels.  grid = (W/32, B*H)
// ---------------------------------------------------------------------------
__global__ void __launch_bounds__(256)
dcn_wmma(const float* __restrict__ xin, const float* __restrict__ om,
         const unsigned short* __restrict__ Wp, const float* __restrict__ bias,
         float* __restrict__ out) {
  constexpr int KST = 584;                        // 576 + 8 pad, 16B-aligned
  __shared__ unsigned short vlds[32 * KST];
  const int tid  = threadIdx.x;
  const int lane = tid & 31, wave = tid >> 5;
  const int lh = lane & 15, lq = lane >> 4;
  const int mw = wave >> 1, nw = wave & 1;
  const int xseg = blockIdx.x * 32;
  const int b = blockIdx.y >> 7, y = blockIdx.y & 127;

  constexpr int K = 576;
  const unsigned short* Arow = Wp + (size_t)(mw * 16 + lh) * K;
  __builtin_prefetch(Arow, 0, 1);                 // global_prefetch_b8

  // phase 1: modulated bilinear sampling -> LDS (offset ch c == om ch c)
  const size_t omb = (size_t)b * 216 * HH * WW;
  for (int i = tid; i < GG * 9 * 32; i += 256) {  // 2304 = 9*256 exact
    int xl = i & 31; int t = i >> 5; int kk = t % 9; int g = t / 9;
    int xg = xseg + xl;
    size_t pix = (size_t)y * WW + xg;
    float dy = om[omb + (size_t)(g * 18 + kk * 2)     * (HH * WW) + pix];
    float dx = om[omb + (size_t)(g * 18 + kk * 2 + 1) * (HH * WW) + pix];
    float mr = om[omb + (size_t)(144 + g * 9 + kk)    * (HH * WW) + pix];
    float m = 1.f / (1.f + __expf(-mr));
    int ky = kk / 3, kx = kk % 3;
    float py = (float)(y - 1 + ky) + dy;
    float px = (float)(xg - 1 + kx) + dx;
    float fy = floorf(py), fx = floorf(px);
    int y0 = (int)fy, x0 = (int)fx;
    float ly = py - fy, lx = px - fx;
    bool iy0 = (y0 >= 0) && (y0 < HH), iy1 = (y0 + 1 >= 0) && (y0 + 1 < HH);
    bool ix0 = (x0 >= 0) && (x0 < WW), ix1 = (x0 + 1 >= 0) && (x0 + 1 < WW);
    float a00 = (iy0 && ix0) ? (1.f - ly) * (1.f - lx) : 0.f;
    float a01 = (iy0 && ix1) ? (1.f - ly) * lx         : 0.f;
    float a10 = (iy1 && ix0) ? ly * (1.f - lx)         : 0.f;
    float a11 = (iy1 && ix1) ? ly * lx                 : 0.f;
    int y0c = iclamp(y0, 0, HH - 1), y1c = iclamp(y0 + 1, 0, HH - 1);
    int x0c = iclamp(x0, 0, WW - 1), x1c = iclamp(x0 + 1, 0, WW - 1);
    unsigned int pk[4];
#pragma unroll
    for (int c = 0; c < 8; c += 2) {
      const float* p0 = xin + ((size_t)b * NF + g * 8 + c) * (HH * WW);
      const float* p1 = p0 + (HH * WW);
      float s0 = a00 * p0[y0c * WW + x0c] + a01 * p0[y0c * WW + x1c]
               + a10 * p0[y1c * WW + x0c] + a11 * p0[y1c * WW + x1c];
      float s1 = a00 * p1[y0c * WW + x0c] + a01 * p1[y0c * WW + x1c]
               + a10 * p1[y1c * WW + x0c] + a11 * p1[y1c * WW + x1c];
      pk[c >> 1] = (unsigned int)f2bf(s0 * m) | ((unsigned int)f2bf(s1 * m) << 16);
    }
    uint4 st; st.x = pk[0]; st.y = pk[1]; st.z = pk[2]; st.w = pk[3];
    *(uint4*)(&vlds[xl * KST + kk * 64 + g * 8]) = st;   // ds_store_b128
  }
  __syncthreads();

  // phase 2: D = W(64x576) * V(576x32), 18 WMMA steps, b128 B loads
  const int xcol = nw * 16 + lh;
  const unsigned short* brow = &vlds[xcol * KST + lq * 16];
  v8f acc = {0.f, 0.f, 0.f, 0.f, 0.f, 0.f, 0.f, 0.f};
#pragma unroll
  for (int kt = 0; kt < 18; ++kt) {
    Frag af, bfr;
    const int kb0 = kt * 32;
#pragma unroll
    for (int i = 0; i < 8; ++i) {
      int ka = kb0 + lq * 16 + i * 2;
      af.u[i] = *(const unsigned int*)(Arow + ka);
    }
    const unsigned short* bp = brow + kb0;
    bfr.q[0] = *(const uint4*)(bp);
    bfr.q[1] = *(const uint4*)(bp + 8);
    acc = __builtin_amdgcn_wmma_f32_16x16x32_bf16(false, af.v, false, bfr.v,
                                                  (short)0, acc, false, false);
  }

  const int colx = xseg + xcol;
#pragma unroll
  for (int r = 0; r < 8; ++r) {
    int co = mw * 16 + r + lq * 8;
    float v = acc[r] + bias[co];
    v = (v >= 0.f) ? v : 0.1f * v;                 // lrelu(dcn(...))
    out[(((size_t)b * NF + co) * HH + y) * WW + colx] = v;
  }
}

// ---------------------------------------------------------------------------
// 1x1 conv over concat(ref_, nbr_) : Out(64) = W(64x128) * X(128) + b.
// LDS tile is pixel-major [pix][ch] so B fragments are b128 loads.
// WG = 8 waves = 4(M) x 2(N) -> 64 Cout x 32 pixels.  grid = (H*W/32, B)
// ---------------------------------------------------------------------------
__global__ void __launch_bounds__(256)
conv1x1_wmma(const float* __restrict__ ref_, const float* __restrict__ nbr_,
             const unsigned short* __restrict__ Wp, const float* __restrict__ bias,
             float* __restrict__ out) {
  __shared__ unsigned short flds[32 * 128];
  const int tid  = threadIdx.x;
  const int lane = tid & 31, wave = tid >> 5;
  const int lh = lane & 15, lq = lane >> 4;
  const int mw = wave >> 1, nw = wave & 1;
  const int pseg = blockIdx.x * 32;
  const int b = blockIdx.y;
  const int HWp = HH * WW;

  const unsigned short* Arow = Wp + (size_t)(mw * 16 + lh) * 128;
  __builtin_prefetch(Arow, 0, 1);                  // global_prefetch_b8

  for (int i = tid; i < 128 * 32; i += 256) {      // 4096 = 16*256 exact
    int xl = i & 31; int ch = i >> 5;
    float v = (ch < 64) ? ref_[((size_t)b * 64 + ch) * HWp + pseg + xl]
                        : nbr_[((size_t)b * 64 + (ch - 64)) * HWp + pseg + xl];
    flds[xl * 128 + ch] = f2bf(v);
  }
  __syncthreads();

  const int xcol = nw * 16 + lh;
  const unsigned short* brow = &flds[xcol * 128 + lq * 16];
  v8f acc = {0.f, 0.f, 0.f, 0.f, 0.f, 0.f, 0.f, 0.f};
#pragma unroll
  for (int kt = 0; kt < 4; ++kt) {
    Frag af, bfr;
#pragma unroll
    for (int i = 0; i < 8; ++i) {
      int ka = kt * 32 + lq * 16 + i * 2;
      af.u[i] = *(const unsigned int*)(Arow + ka);
    }
    const unsigned short* bp = brow + kt * 32;
    bfr.q[0] = *(const uint4*)(bp);
    bfr.q[1] = *(const uint4*)(bp + 8);
    acc = __builtin_amdgcn_wmma_f32_16x16x32_bf16(false, af.v, false, bfr.v,
                                                  (short)0, acc, false, false);
  }

#pragma unroll
  for (int r = 0; r < 8; ++r) {
    int co = mw * 16 + r + lq * 8;
    out[((size_t)b * 64 + co) * HWp + pseg + xcol] = acc[r] + bias[co];
  }
}

// ---------------------------------------------------------------------------
extern "C" void kernel_launch(void* const* d_in, const int* in_sizes, int n_in,
                              void* d_out, int out_size, void* d_ws, size_t ws_size,
                              hipStream_t stream) {
  const float* nbr    = (const float*)d_in[0];
  const float* ref    = (const float*)d_in[1];
  const float* W_oc1  = (const float*)d_in[2];  const float* b_oc1  = (const float*)d_in[3];
  const float* W_oc2  = (const float*)d_in[4];  const float* b_oc2  = (const float*)d_in[5];
  const float* W_om1  = (const float*)d_in[6];  const float* b_om1  = (const float*)d_in[7];
  const float* W_dcn1 = (const float*)d_in[8];  const float* b_dcn1 = (const float*)d_in[9];
  const float* W_oc3  = (const float*)d_in[10]; const float* b_oc3  = (const float*)d_in[11];
  const float* W_oc4  = (const float*)d_in[12]; const float* b_oc4  = (const float*)d_in[13];
  const float* W_om2  = (const float*)d_in[14]; const float* b_om2  = (const float*)d_in[15];
  const float* W_dcn2 = (const float*)d_in[16]; const float* b_dcn2 = (const float*)d_in[17];
  const float* W_1x1  = (const float*)d_in[18]; const float* b_1x1  = (const float*)d_in[19];

  char* ws = (char*)d_ws;
  size_t cur = 0;
  auto alloc = [&](size_t bytes) {
    void* p = ws + cur;
    cur += (bytes + 255) & ~(size_t)255;
    return p;
  };
  const size_t featB = (size_t)BB * NF * HH * WW * sizeof(float);      // 16.8 MB
  float* off_a = (float*)alloc(featB);
  float* off_b = (float*)alloc(featB);
  float* om    = (float*)alloc((size_t)BB * 216 * HH * WW * sizeof(float)); // 56.6 MB
  float* nbr_o = (float*)alloc(featB);
  float* ref_o = (float*)alloc(featB);
  unsigned short* Woc1p  = (unsigned short*)alloc((size_t)64  * 1152 * 2);
  unsigned short* Woc2p  = (unsigned short*)alloc((size_t)64  * 576  * 2);
  unsigned short* Wom1p  = (unsigned short*)alloc((size_t)224 * 576  * 2);
  unsigned short* Wdcn1p = (unsigned short*)alloc((size_t)64  * 576  * 2);
  unsigned short* Woc3p  = (unsigned short*)alloc((size_t)64  * 1152 * 2);
  unsigned short* Woc4p  = (unsigned short*)alloc((size_t)64  * 576  * 2);
  unsigned short* Wom2p  = (unsigned short*)alloc((size_t)224 * 576  * 2);
  unsigned short* Wdcn2p = (unsigned short*)alloc((size_t)64  * 576  * 2);
  unsigned short* W1x1p  = (unsigned short*)alloc((size_t)64  * 128  * 2);

  auto pack = [&](const float* w, unsigned short* wp, int Cout, int CIN, int KK, int Mpad) {
    int total = Mpad * CIN * KK;
    pack_weights<<<(total + 255) / 256, 256, 0, stream>>>(w, wp, Cout, CIN, KK, Mpad);
  };
  pack(W_oc1,  Woc1p,  64,  128, 9, 64);
  pack(W_oc2,  Woc2p,  64,  64,  9, 64);
  pack(W_om1,  Wom1p,  216, 64,  9, 224);
  pack(W_dcn1, Wdcn1p, 64,  64,  9, 64);
  pack(W_oc3,  Woc3p,  64,  128, 9, 64);
  pack(W_oc4,  Woc4p,  64,  64,  9, 64);
  pack(W_om2,  Wom2p,  216, 64,  9, 224);
  pack(W_dcn2, Wdcn2p, 64,  64,  9, 64);
  pack(W_1x1,  W1x1p,  64,  128, 1, 64);

  const dim3 blk(256);
  const dim3 gconv64(WW / 64, BB * HH, 2);   // Cout=64  -> 4 M-tiles / 2 per WG
  const dim3 gconv216(WW / 64, BB * HH, 7);  // Mpad=224 -> 14 M-tiles / 2 per WG
  const dim3 gdcn(WW / 32, BB * HH);
  const dim3 g1x1(HH * WW / 32, BB);

  // ---- branch 1: nbr ----
  conv3x3_wmma<128><<<gconv64,  blk, 0, stream>>>(nbr,  ref,  64, Woc1p, b_oc1, 64,  off_a, 1);
  conv3x3_wmma<64><<<gconv64,  blk, 0, stream>>>(off_a, off_a, 64, Woc2p, b_oc2, 64,  off_b, 1);
  conv3x3_wmma<64><<<gconv216, blk, 0, stream>>>(off_b, off_b, 64, Wom1p, b_om1, 216, om,    0);
  dcn_wmma<<<gdcn, blk, 0, stream>>>(nbr, om, Wdcn1p, b_dcn1, nbr_o);

  // ---- branch 2: ref (reuse off_a/off_b/om) ----
  conv3x3_wmma<128><<<gconv64,  blk, 0, stream>>>(nbr,  ref,  64, Woc3p, b_oc3, 64,  off_a, 1);
  conv3x3_wmma<64><<<gconv64,  blk, 0, stream>>>(off_a, off_a, 64, Woc4p, b_oc4, 64,  off_b, 1);
  conv3x3_wmma<64><<<gconv216, blk, 0, stream>>>(off_b, off_b, 64, Wom2p, b_om2, 216, om,    0);
  dcn_wmma<<<gdcn, blk, 0, stream>>>(ref, om, Wdcn2p, b_dcn2, ref_o);

  // ---- fuse: conv1x1(concat(ref_, nbr_)) ----
  conv1x1_wmma<<<g1x1, blk, 0, stream>>>(ref_o, nbr_o, W1x1p, b_1x1, (float*)d_out);
}
